// My_Mamba_49014166782184
// MI455X (gfx1250) — compile-verified, matrix operand order checked by reading, not run on hardware
//
#include <hip/hip_runtime.h>
#include <math.h>

// ---------------------------------------------------------------------------
// CDNA5 / gfx1250 wave32 WMMA types
// ---------------------------------------------------------------------------
typedef __attribute__((ext_vector_type(16))) __bf16 v16bf;
typedef __attribute__((ext_vector_type(8)))  float  v8f;

// ---------------------------------------------------------------------------
// Generic GEMM:  C[M,N] = act(A[M,K] @ B[K,N] + bias[N]) (+ Res[M,N])
// fp32 in memory, converted to bf16 in registers -> v_wmma_f32_16x16x32_bf16.
// One wave: 64 rows x 16 cols (4 accumulator tiles, B fragment reused 4x).
// Weight matrix streams from HBM once; A re-reads hit L2 (192 MB).
// M/N edges are handled by ADDRESS CLAMPING (dead rows/cols compute garbage
// that is never stored), so the vectorized unpredicated path needs only
// K % 32 == 0.  Guarded path (K tail) zero-selects only B's k-overflow.
// k-loop unrolled 2x so the scheduler can fill the WMMA->VALU hazard window
// (4 co-exec slots for bf16 WMMA) with the other iteration's loads/converts.
// ACT: 0=none 1=relu 2=softplus
// ---------------------------------------------------------------------------
template <int ACT, bool HAS_BIAS, bool HAS_RES>
__global__ __launch_bounds__(128)
void gemm_bf16_wmma(const float* __restrict__ A, int lda,
                    const float* __restrict__ Bw, int ldb,
                    const float* __restrict__ bias,
                    const float* Res, int ldres,
                    float* C, int ldc,
                    int M, int N, int K)
{
    const int lane  = threadIdx.x & 31;
    const int wave  = threadIdx.x >> 5;
    const int col0  = (blockIdx.x * 4 + wave) * 16;  // 16 cols per wave
    if (col0 >= N) return;                           // uniform: dead wave
    const int row0  = blockIdx.y * 64;               // 64 rows per block
    const int half  = lane >> 4;                     // k-half select
    const int lidx  = lane & 15;                     // M row (A) / N col (B)
    const int n     = col0 + lidx;

    const int mmax = M - 1;
    const int nc   = (n < N) ? n : (N - 1);          // clamped B column

    v8f acc[4] = {};

    if ((K & 31) == 0) {
        // ---------------- fast path: no per-element predication ------------
        const size_t bstep = (size_t)32 * ldb;
        const float* bp = Bw + (size_t)(half * 8 * ldb + nc);
        const float* ap[4];
#pragma unroll
        for (int t = 0; t < 4; ++t) {
            int m = row0 + t * 16 + lidx;
            if (m > mmax) m = mmax;                  // clamp: garbage rows never stored
            ap[t] = A + (size_t)(m * lda + half * 8);
        }

#pragma unroll 2
        for (int kb = 0; kb < K; kb += 32) {
            if (kb + 64 < K)                          // prefetch B two k-slabs ahead
                __builtin_prefetch(bp + 2 * bstep, 0, 1);

            v16bf bf;
#pragma unroll
            for (int j = 0; j < 8; ++j)  bf[j]     = (__bf16)bp[(size_t)j * ldb];
#pragma unroll
            for (int j = 0; j < 8; ++j)  bf[j + 8] = (__bf16)bp[(size_t)(j + 16) * ldb];

#pragma unroll
            for (int t = 0; t < 4; ++t) {
                const float4 a0 = *(const float4*)(ap[t] + 0);
                const float4 a1 = *(const float4*)(ap[t] + 4);
                const float4 a2 = *(const float4*)(ap[t] + 16);
                const float4 a3 = *(const float4*)(ap[t] + 20);
                v16bf af;
                af[0]  = (__bf16)a0.x; af[1]  = (__bf16)a0.y; af[2]  = (__bf16)a0.z; af[3]  = (__bf16)a0.w;
                af[4]  = (__bf16)a1.x; af[5]  = (__bf16)a1.y; af[6]  = (__bf16)a1.z; af[7]  = (__bf16)a1.w;
                af[8]  = (__bf16)a2.x; af[9]  = (__bf16)a2.y; af[10] = (__bf16)a2.z; af[11] = (__bf16)a2.w;
                af[12] = (__bf16)a3.x; af[13] = (__bf16)a3.y; af[14] = (__bf16)a3.z; af[15] = (__bf16)a3.w;
                acc[t] = __builtin_amdgcn_wmma_f32_16x16x32_bf16(
                    false, af, false, bf, (short)0, acc[t], false, false);
                ap[t] += 32;
            }
            bp += bstep;
        }
    } else {
        // ------------- guarded path (K not multiple of 32) ------------------
        // All addresses clamped (always valid). Only B's k-overflow elements
        // are zero-selected; that zeroes the products of A's garbage too.
        const int kmax = K - 1;
        for (int kb = 0; kb < K; kb += 32) {
            float tb[16];
#pragma unroll
            for (int j = 0; j < 16; ++j) {
                int k = kb + ((j < 8) ? j : j + 8) + 8 * half;
                if (k > kmax) k = kmax;
                tb[j] = Bw[(size_t)(k * ldb + nc)];
            }
            v16bf bf;
#pragma unroll
            for (int j = 0; j < 16; ++j) {
                int k = kb + ((j < 8) ? j : j + 8) + 8 * half;
                bf[j] = (__bf16)((k < K) ? tb[j] : 0.0f);
            }
#pragma unroll
            for (int t = 0; t < 4; ++t) {
                int m = row0 + t * 16 + lidx;
                if (m > mmax) m = mmax;
                float ta[16];
#pragma unroll
                for (int j = 0; j < 16; ++j) {
                    int k = kb + ((j < 8) ? j : j + 8) + 8 * half;
                    if (k > kmax) k = kmax;
                    ta[j] = A[(size_t)(m * lda + k)];
                }
                v16bf af;
#pragma unroll
                for (int j = 0; j < 16; ++j) af[j] = (__bf16)ta[j];
                acc[t] = __builtin_amdgcn_wmma_f32_16x16x32_bf16(
                    false, af, false, bf, (short)0, acc[t], false, false);
            }
        }
    }

    // ---- epilogue: C/D layout row = base + r + 8*half, col = lidx ----------
    if (n < N) {
        float bv = HAS_BIAS ? bias[n] : 0.0f;
#pragma unroll
        for (int t = 0; t < 4; ++t) {
            if (row0 + t * 16 >= M) continue;            // uniform
#pragma unroll
            for (int r = 0; r < 8; ++r) {
                int row = row0 + t * 16 + r + 8 * half;
                if (row < M) {
                    float v = acc[t][r] + bv;
                    if (ACT == 1) v = fmaxf(v, 0.0f);
                    else if (ACT == 2) v = (v > 20.0f) ? v : log1pf(__expf(v));
                    if (HAS_RES) v += Res[(size_t)(row * ldres + n)];
                    C[(size_t)(row * ldc + n)] = v;
                }
            }
        }
    }
}

// ---------------------------------------------------------------------------
// RMSNorm over rows of D elements: y = x * rsqrt(mean(x^2)+eps) * w
// ---------------------------------------------------------------------------
__global__ __launch_bounds__(256)
void rmsnorm_kernel(const float* __restrict__ X, const float* __restrict__ W,
                    float* __restrict__ Y, int D)
{
    const int row = blockIdx.x;
    const float* x = X + (size_t)row * D;
    float* y = Y + (size_t)row * D;
    __shared__ float red[9];

    float ss = 0.0f;
    for (int d = threadIdx.x; d < D; d += blockDim.x) { float v = x[d]; ss += v * v; }
#pragma unroll
    for (int off = 16; off > 0; off >>= 1) ss += __shfl_down(ss, off, 32);
    if ((threadIdx.x & 31) == 0) red[threadIdx.x >> 5] = ss;
    __syncthreads();
    if (threadIdx.x == 0) {
        float t = 0.0f;
        for (int i = 0; i < (int)(blockDim.x >> 5); ++i) t += red[i];
        red[8] = rsqrtf(t / (float)D + 1e-5f);
    }
    __syncthreads();
    const float scale = red[8];
    for (int d = threadIdx.x; d < D; d += blockDim.x) y[d] = x[d] * scale * W[d];
}

// ---------------------------------------------------------------------------
// Depthwise causal conv (K taps) over L + SiLU.
// proj (B*L, 2I): hs is columns [0,I). Output ut (B*L, I).
// ---------------------------------------------------------------------------
__global__ __launch_bounds__(256)
void conv_silu_kernel(const float* __restrict__ proj,
                      const float* __restrict__ cw,   // (I,K)
                      const float* __restrict__ cb,   // (I)
                      float* __restrict__ ut,
                      int Bn, int L, int I, int K)
{
    int idx = blockIdx.x * blockDim.x + threadIdx.x;
    int total = Bn * L * I;
    if (idx >= total) return;
    int i = idx % I;
    int l = (idx / I) % L;
    int b = idx / (I * L);
    float s = cb[i];
#pragma unroll 4
    for (int k = 0; k < K; ++k) {
        int ls = l + k - (K - 1);
        bool ok = (ls >= 0);
        int off = ok ? ((b * L + ls) * (2 * I) + i) : 0;
        float v = proj[(size_t)off];
        s += cw[i * K + k] * (ok ? v : 0.0f);
    }
    ut[(size_t)idx] = s * (1.0f / (1.0f + __expf(-s)));
}

// ---------------------------------------------------------------------------
// Selective-scan: per (b,i) channel, N=16 states carried in registers across
// L=60 steps; fused +u*D and *silu(gate).
// ---------------------------------------------------------------------------
#define NS 16
__global__ __launch_bounds__(256)
void scan_kernel(const float* __restrict__ ssm,   // (B*L, R+2N)
                 const float* __restrict__ dtl,   // (B*L, I)
                 const float* __restrict__ ut,    // (B*L, I)
                 const float* __restrict__ proj,  // (B*L, 2I)  gate at col I+i
                 const float* __restrict__ alog,  // (I, N)
                 const float* __restrict__ Dp,    // (I)
                 float* __restrict__ y,           // (B*L, I)
                 int Bn, int L, int I, int R)
{
    int idx = blockIdx.x * blockDim.x + threadIdx.x;
    if (idx >= Bn * I) return;
    int i = idx % I;
    int b = idx / I;

    float Ar[NS], s[NS];
#pragma unroll
    for (int n = 0; n < NS; ++n) { Ar[n] = -__expf(alog[(size_t)(i * NS + n)]); s[n] = 0.0f; }
    const float Dv = Dp[i];
    const int ldssm = R + 2 * NS;

    for (int l = 0; l < L; ++l) {
        int row = b * L + l;
        float dt = dtl[(size_t)(row * I + i)];
        float u  = ut[(size_t)(row * I + i)];
        const float* sr = ssm + (size_t)row * ldssm;
        float acc = 0.0f;
#pragma unroll
        for (int n = 0; n < NS; ++n) {
            float dA = __expf(dt * Ar[n]);
            s[n] = dA * s[n] + dt * sr[R + n] * u;
            acc += s[n] * sr[R + NS + n];
        }
        float out = acc + u * Dv;
        float g = proj[(size_t)(row * (2 * I) + I + i)];
        out *= g * (1.0f / (1.0f + __expf(-g)));
        y[(size_t)(row * I + i)] = out;
    }
}

// ---------------------------------------------------------------------------
// Host-side dispatch helpers
// ---------------------------------------------------------------------------
enum { ACT_NONE = 0, ACT_RELU = 1, ACT_SOFTPLUS = 2 };

static void gemm_disp(hipStream_t stream, int act,
                      const float* A, int lda, const float* Bw, int ldb,
                      const float* bias, const float* Res, int ldres,
                      float* C, int ldc, int M, int N, int K)
{
    dim3 grid((N + 63) / 64, (M + 63) / 64), blk(128);
    if (act == ACT_RELU)
        gemm_bf16_wmma<1, true, false><<<grid, blk, 0, stream>>>(A, lda, Bw, ldb, bias, Res, ldres, C, ldc, M, N, K);
    else if (act == ACT_SOFTPLUS)
        gemm_bf16_wmma<2, true, false><<<grid, blk, 0, stream>>>(A, lda, Bw, ldb, bias, Res, ldres, C, ldc, M, N, K);
    else if (Res)
        gemm_bf16_wmma<0, false, true><<<grid, blk, 0, stream>>>(A, lda, Bw, ldb, bias, Res, ldres, C, ldc, M, N, K);
    else if (bias)
        gemm_bf16_wmma<0, true, false><<<grid, blk, 0, stream>>>(A, lda, Bw, ldb, bias, Res, ldres, C, ldc, M, N, K);
    else
        gemm_bf16_wmma<0, false, false><<<grid, blk, 0, stream>>>(A, lda, Bw, ldb, bias, Res, ldres, C, ldc, M, N, K);
}

extern "C" void kernel_launch(void* const* d_in, const int* in_sizes, int n_in,
                              void* d_out, int out_size, void* d_ws, size_t ws_size,
                              hipStream_t stream)
{
    (void)in_sizes; (void)n_in; (void)out_size; (void)ws_size;
    const int Bn = 8, L = 60, F = 13, Dm = 768, I = 1536, N = 16, R = 48, K = 4, NL = 32;
    const int M = Bn * L;            // 480
    const int SSMW = R + 2 * N;      // 80

    const float* x         = (const float*)d_in[0];
    const float* w_in1     = (const float*)d_in[1];
    const float* b_in1     = (const float*)d_in[2];
    const float* w_in2     = (const float*)d_in[3];
    const float* b_in2     = (const float*)d_in[4];
    const float* w_in3     = (const float*)d_in[5];
    const float* b_in3     = (const float*)d_in[6];
    const float* norm_w    = (const float*)d_in[7];
    const float* in_proj_w = (const float*)d_in[8];
    const float* conv_w    = (const float*)d_in[9];
    const float* conv_b    = (const float*)d_in[10];
    const float* x_proj_w  = (const float*)d_in[11];
    const float* dt_proj_w = (const float*)d_in[12];
    const float* dt_proj_b = (const float*)d_in[13];
    const float* A_log     = (const float*)d_in[14];
    const float* D_param   = (const float*)d_in[15];
    const float* out_proj_w= (const float*)d_in[16];
    const float* norm_f_w  = (const float*)d_in[17];
    const float* w_o1      = (const float*)d_in[18];
    const float* b_o1      = (const float*)d_in[19];
    const float* w_o2      = (const float*)d_in[20];
    const float* b_o2      = (const float*)d_in[21];
    const float* w_o3      = (const float*)d_in[22];
    const float* b_o3      = (const float*)d_in[23];

    // workspace carve-up (~19 MB fp32)
    float* ws   = (float*)d_ws;
    float* h    = ws;  ws += (size_t)M * Dm;
    float* hn   = ws;  ws += (size_t)M * Dm;
    float* proj = ws;  ws += (size_t)M * 2 * I;
    float* ut   = ws;  ws += (size_t)M * I;
    float* ssm  = ws;  ws += (size_t)M * SSMW;
    float* dtl  = ws;  ws += (size_t)M * I;
    float* yv   = ws;  ws += (size_t)M * I;
    float* t1   = ws;  ws += (size_t)M * 256;
    float* t2   = ws;  ws += (size_t)M * 256;
    float* o1   = ws;  ws += (size_t)Bn * 256;
    float* o2   = ws;  ws += (size_t)Bn * 256;

    // ---- input MLP: (B*L,13) -> 256 -> 256 -> 768
    gemm_disp(stream, ACT_RELU, x,  F,   w_in1, 256, b_in1, nullptr, 0, t1, 256, M, 256, F);
    gemm_disp(stream, ACT_RELU, t1, 256, w_in2, 256, b_in2, nullptr, 0, t2, 256, M, 256, 256);
    gemm_disp(stream, ACT_NONE, t2, 256, w_in3, Dm,  b_in3, nullptr, 0, h,  Dm,  M, Dm, 256);

    // ---- 32 Mamba layers
    for (int l = 0; l < NL; ++l) {
        const float* nw  = norm_w     + (size_t)l * Dm;
        const float* ipw = in_proj_w  + (size_t)l * Dm * 2 * I;
        const float* cw  = conv_w     + (size_t)l * I * K;
        const float* cb  = conv_b     + (size_t)l * I;
        const float* xpw = x_proj_w   + (size_t)l * I * SSMW;
        const float* dpw = dt_proj_w  + (size_t)l * R * I;
        const float* dpb = dt_proj_b  + (size_t)l * I;
        const float* al  = A_log      + (size_t)l * I * N;
        const float* dp  = D_param    + (size_t)l * I;
        const float* opw = out_proj_w + (size_t)l * I * Dm;

        rmsnorm_kernel<<<M, 256, 0, stream>>>(h, nw, hn, Dm);
        // proj = hn @ in_proj_w   (480x768 @ 768x3072)
        gemm_disp(stream, ACT_NONE, hn, Dm, ipw, 2 * I, nullptr, nullptr, 0, proj, 2 * I, M, 2 * I, Dm);
        // depthwise causal conv + SiLU -> ut (B*L, I)
        {
            int total = Bn * L * I;
            conv_silu_kernel<<<(total + 255) / 256, 256, 0, stream>>>(proj, cw, cb, ut, Bn, L, I, K);
        }
        // ssm = ut @ x_proj_w  (480x1536 @ 1536x80)
        gemm_disp(stream, ACT_NONE, ut, I, xpw, SSMW, nullptr, nullptr, 0, ssm, SSMW, M, SSMW, I);
        // dt = softplus(ssm[:, :R] @ dt_proj_w + dt_proj_b)  (480x48 @ 48x1536)
        gemm_disp(stream, ACT_SOFTPLUS, ssm, SSMW, dpw, I, dpb, nullptr, 0, dtl, I, M, I, R);
        // selective scan + gate -> yv
        {
            int total = Bn * I;
            scan_kernel<<<(total + 255) / 256, 256, 0, stream>>>(ssm, dtl, ut, proj, al, dp, yv, Bn, L, I, R);
        }
        // h = h + yv @ out_proj_w  (480x1536 @ 1536x768), residual fused
        gemm_disp(stream, ACT_NONE, yv, I, opw, Dm, nullptr, h, Dm, h, Dm, M, Dm, I);
    }

    // ---- final norm + output MLP (f = hn viewed as (8, 46080))
    rmsnorm_kernel<<<M, 256, 0, stream>>>(h, norm_f_w, hn, Dm);
    gemm_disp(stream, ACT_RELU, hn, L * Dm, w_o1, 256, b_o1, nullptr, 0, o1, 256, Bn, 256, L * Dm);
    gemm_disp(stream, ACT_RELU, o1, 256,    w_o2, 256, b_o2, nullptr, 0, o2, 256, Bn, 256, 256);
    gemm_disp(stream, ACT_NONE, o2, 256,    w_o3, 1,   b_o3, nullptr, 0, (float*)d_out, 1, Bn, 1, 256);
}
